// TimeEquivConvLayer_81406810128547
// MI455X (gfx1250) — matrix operand order repaired and checked by reading, not compiled, not run on hardware
//
#include <hip/hip_runtime.h>
#include <hip/hip_bf16.h>
#include <stdint.h>

typedef __attribute__((ext_vector_type(16))) _Float16 v16h;
typedef __attribute__((ext_vector_type(8)))  float    v8f;

#define T_DIM   8
#define N_NODES 10000
#define E_EDGES 40000
#define S_INc   16
#define V_INc   8
#define S_OUTc  16
#define V_OUTc  8
#define C_IN    40
#define C_OUT   40
#define HID     64
#define WNW     576

#define C1c 0.25f            /* 1/sqrt(16) */
#define C2c 0.25f            /* 1/sqrt(16) */
#define C3c 0.35355339059f   /* 1/sqrt(8)  */
#define C4c 0.20412414523f   /* 1/sqrt(24) */
#define BN_EPS 1e-5f

// ---------------------------------------------------------------------------
// Weight f32 -> f16 conversion, pre-packed into per-lane WMMA B-fragment
// order: frag_base = ((ntile*2 + kstep)*32 + lane)*16 + j, where
//   k = kstep*32 + (j>>3)*16 + (lane>>4)*8 + (j&7),  n = ntile*16 + (lane&15)
// so each lane's 16-element fragment is one contiguous 32-byte load.
// ---------------------------------------------------------------------------
__global__ void prep_weights(const float* __restrict__ w1, const float* __restrict__ w2,
                             _Float16* __restrict__ w1p, _Float16* __restrict__ w2p) {
    int idx = blockIdx.x * 256 + threadIdx.x;
    if (idx < 4 * 2 * 32 * 16) {                 // W1: 4 N-tiles
        int j = idx & 15, lane = (idx >> 4) & 31, ks = (idx >> 9) & 1, n0 = idx >> 10;
        int k = ks * 32 + (j >> 3) * 16 + (lane >> 4) * 8 + (j & 7);
        int n = n0 * 16 + (lane & 15);
        w1p[idx] = (_Float16)w1[k * HID + n];
    }
    if (idx < 36 * 2 * 32 * 16) {                // W2: 36 N-tiles
        int j = idx & 15, lane = (idx >> 4) & 31, ks = (idx >> 9) & 1, nt = idx >> 10;
        int k = ks * 32 + (j >> 3) * 16 + (lane >> 4) * 8 + (j & 7);
        int n = nt * 16 + (lane & 15);
        w2p[idx] = (_Float16)w2[k * WNW + n];
    }
}

// ---------------------------------------------------------------------------
// Spectral conv: rfft(T=8) -> keep modes 0,1 -> complex weight -> irfft.
// irfft uses only Re of bin0:  x[t] = (1/8)(ReY0 + 2(ReY1 cos - ImY1 sin))
// ---------------------------------------------------------------------------
__global__ void spectral_kernel(const float* __restrict__ x,
                                const float* __restrict__ wr,
                                const float* __restrict__ wi,
                                float* __restrict__ xsc) {
    __shared__ float X0[16][16], X1r[16][16], X1i[16][16];
    const float COS8[8] = {1.f, 0.70710678f, 0.f, -0.70710678f, -1.f, -0.70710678f, 0.f, 0.70710678f};
    const float SIN8[8] = {0.f, 0.70710678f, 1.f, 0.70710678f, 0.f, -0.70710678f, -1.f, -0.70710678f};
    int ln = threadIdx.x & 15;   // phase1: i, phase2: o
    int nl = threadIdx.x >> 4;   // local node
    int n  = blockIdx.x * 16 + nl;

    float a0 = 0.f, a1r = 0.f, a1i = 0.f;
#pragma unroll
    for (int t = 0; t < 8; ++t) {
        float v = x[((size_t)t * N_NODES + n) * C_IN + ln];
        a0  += v;
        a1r += v * COS8[t];
        a1i -= v * SIN8[t];      // e^{-i 2 pi t / 8}
    }
    X0[nl][ln] = a0; X1r[nl][ln] = a1r; X1i[nl][ln] = a1i;
    __syncthreads();

    float y0r = 0.f, y1r = 0.f, y1i = 0.f;
#pragma unroll
    for (int i = 0; i < 16; ++i) {
        float wr0 = wr[(i * 16 + ln) * 2 + 0];
        float wr1 = wr[(i * 16 + ln) * 2 + 1];
        float wi1 = wi[(i * 16 + ln) * 2 + 1];
        y0r += X0[nl][i] * wr0;
        y1r += X1r[nl][i] * wr1 - X1i[nl][i] * wi1;
        y1i += X1r[nl][i] * wi1 + X1i[nl][i] * wr1;
    }
#pragma unroll
    for (int t = 0; t < 8; ++t)
        xsc[((size_t)t * N_NODES + n) * S_OUTc + ln] =
            0.125f * (y0r + 2.f * (y1r * COS8[t] - y1i * SIN8[t]));
}

// ---------------------------------------------------------------------------
// In-degree counts -> reciprocal of clip(count,1)
// ---------------------------------------------------------------------------
__global__ void count_kernel(const int* __restrict__ ei, float* __restrict__ cnt) {
    int e = blockIdx.x * 256 + threadIdx.x;
    if (e < E_EDGES) atomicAdd(&cnt[ei[E_EDGES + e]], 1.0f);
}
__global__ void invc_kernel(float* __restrict__ cnt) {
    int n = blockIdx.x * 256 + threadIdx.x;
    if (n < N_NODES) cnt[n] = 1.0f / fmaxf(cnt[n], 1.0f);
}

// ---------------------------------------------------------------------------
// Fused edge kernel: per-wave 32-row tile (32 edges at one t).
// B fragments (prepacked, 2x b128 per frag) are reused across the two 16-row
// A tiles (M-blocking): GEMM1 = 16 WMMAs, GEMM2 = 144 WMMAs per wave.
// ---------------------------------------------------------------------------
__global__ __launch_bounds__(64) void edge_kernel(
    const float* __restrict__ x,   const int* __restrict__ ei,
    const float* __restrict__ ea,  const float* __restrict__ esh,
    const float* __restrict__ b1,  const float* __restrict__ b2,
    const _Float16* __restrict__ w1p, const _Float16* __restrict__ w2p,
    float* __restrict__ sums)
{
    const int w    = threadIdx.x >> 5;             // wave in block (0..1)
    const int lane = threadIdx.x & 31;
    const int half = lane >> 4;
    const int nc   = lane & 15;
    const int tile = blockIdx.x * 2 + w;           // 0..9999
    const int TILES_PER_T = E_EDGES / 32;          // 1250
    const int t      = tile / TILES_PER_T;
    const int e_base = (tile % TILES_PER_T) * 32;

    __shared__ _Float16 sh_hA[2][32][64];
    __shared__ float sh_ew[2][32][16];
    __shared__ float sh_msg[2][32][40];
    __shared__ float sh_s0[2][32];
    __shared__ float sh_s1[2][32][3];
    __shared__ float sh_xs[2][32][16];
    __shared__ float sh_xv[2][32][24];
    __shared__ float sh_vd[2][32][8];
    __shared__ int   sh_dst[2][32];

    // prefetch next tile's A rows (global_prefetch_b8)
    __builtin_prefetch(ea + ((size_t)t * E_EDGES + ((e_base + 32) % E_EDGES)) * HID, 0, 1);

    // ---- stage per-row scalars: each lane owns one of the 32 rows
    {
        int e = e_base + lane;
        int srcn = ei[e];
        sh_dst[w][lane] = ei[E_EDGES + e];
        const float* xr = x + ((size_t)t * N_NODES + (size_t)srcn) * C_IN;
#pragma unroll
        for (int i = 0; i < 16; ++i) sh_xs[w][lane][i] = xr[i];
#pragma unroll
        for (int i = 0; i < 24; ++i) sh_xv[w][lane][i] = xr[16 + i];
        const float* sp = esh + ((size_t)t * E_EDGES + e) * 4;
        float s1x = sp[1], s1y = sp[2], s1z = sp[3];
        sh_s0[w][lane] = sp[0];
        sh_s1[w][lane][0] = s1x; sh_s1[w][lane][1] = s1y; sh_s1[w][lane][2] = s1z;
#pragma unroll
        for (int i = 0; i < 8; ++i)
            sh_vd[w][lane][i] = xr[16+i*3]*s1x + xr[16+i*3+1]*s1y + xr[16+i*3+2]*s1z;
#pragma unroll
        for (int c = 0; c < 40; ++c) sh_msg[w][lane][c] = 0.f;
    }
    __builtin_amdgcn_wave_barrier();

    // ---- GEMM1 A-fragments (f32 global -> f16), rows nc (lo) and 16+nc (hi)
    const float* arow_lo = ea + ((size_t)t * E_EDGES + e_base + nc) * HID;
    const float* arow_hi = arow_lo + 16 * HID;
    v16h a0l, a1l, a0h, a1h;
#pragma unroll
    for (int j = 0; j < 8; ++j) {
        a0l[j]     = (_Float16)arow_lo[half*8 + j];
        a0l[j + 8] = (_Float16)arow_lo[16 + half*8 + j];
        a1l[j]     = (_Float16)arow_lo[32 + half*8 + j];
        a1l[j + 8] = (_Float16)arow_lo[48 + half*8 + j];
        a0h[j]     = (_Float16)arow_hi[half*8 + j];
        a0h[j + 8] = (_Float16)arow_hi[16 + half*8 + j];
        a1h[j]     = (_Float16)arow_hi[32 + half*8 + j];
        a1h[j + 8] = (_Float16)arow_hi[48 + half*8 + j];
    }

    // ---- GEMM1: 4 N-tiles x 2 K-steps x 2 M-tiles
#pragma unroll
    for (int n0 = 0; n0 < 4; ++n0) {
        int ncol = n0 * 16 + nc;
        float bv = b1[ncol];
        v8f accl, acch;
#pragma unroll
        for (int r = 0; r < 8; ++r) { accl[r] = bv; acch[r] = bv; }
        v16h bfr = *(const v16h*)(w1p + ((n0*2 + 0)*32 + lane)*16);
        accl = __builtin_amdgcn_wmma_f32_16x16x32_f16(false, a0l, false, bfr, (short)0, accl, false, false);
        acch = __builtin_amdgcn_wmma_f32_16x16x32_f16(false, a0h, false, bfr, (short)0, acch, false, false);
        bfr = *(const v16h*)(w1p + ((n0*2 + 1)*32 + lane)*16);
        accl = __builtin_amdgcn_wmma_f32_16x16x32_f16(false, a1l, false, bfr, (short)0, accl, false, false);
        acch = __builtin_amdgcn_wmma_f32_16x16x32_f16(false, a1h, false, bfr, (short)0, acch, false, false);
#pragma unroll
        for (int r = 0; r < 8; ++r) {
            sh_hA[w][half*8 + r][ncol]      = (_Float16)fmaxf(accl[r], 0.f);  // bias+relu
            sh_hA[w][16 + half*8 + r][ncol] = (_Float16)fmaxf(acch[r], 0.f);
        }
    }
    __builtin_amdgcn_wave_barrier();

    // ---- D->A relayout through LDS for GEMM2
    v16h h0l, h1l, h0h, h1h;
#pragma unroll
    for (int j = 0; j < 8; ++j) {
        h0l[j]     = sh_hA[w][nc][half*8 + j];
        h0l[j + 8] = sh_hA[w][nc][16 + half*8 + j];
        h1l[j]     = sh_hA[w][nc][32 + half*8 + j];
        h1l[j + 8] = sh_hA[w][nc][48 + half*8 + j];
        h0h[j]     = sh_hA[w][16 + nc][half*8 + j];
        h0h[j + 8] = sh_hA[w][16 + nc][16 + half*8 + j];
        h1h[j]     = sh_hA[w][16 + nc][32 + half*8 + j];
        h1h[j + 8] = sh_hA[w][16 + nc][48 + half*8 + j];
    }

    // ---- GEMM2 (36 N-tiles) fused with message contraction
    for (int nt = 0; nt < 36; ++nt) {
        int wcol = nt * 16 + nc;
        float bv = b2[wcol];
        v8f accl, acch;
#pragma unroll
        for (int r = 0; r < 8; ++r) { accl[r] = bv; acch[r] = bv; }
        v16h bfr = *(const v16h*)(w2p + ((nt*2 + 0)*32 + lane)*16);
        accl = __builtin_amdgcn_wmma_f32_16x16x32_f16(false, h0l, false, bfr, (short)0, accl, false, false);
        acch = __builtin_amdgcn_wmma_f32_16x16x32_f16(false, h0h, false, bfr, (short)0, acch, false, false);
        bfr = *(const v16h*)(w2p + ((nt*2 + 1)*32 + lane)*16);
        accl = __builtin_amdgcn_wmma_f32_16x16x32_f16(false, h1l, false, bfr, (short)0, accl, false, false);
        acch = __builtin_amdgcn_wmma_f32_16x16x32_f16(false, h1h, false, bfr, (short)0, acch, false, false);
#pragma unroll
        for (int r = 0; r < 8; ++r) {
            sh_ew[w][half*8 + r][nc]      = accl[r];
            sh_ew[w][16 + half*8 + r][nc] = acch[r];
        }
        __builtin_amdgcn_wave_barrier();

        // decode ew columns -> equivariant message contributions (ds_add_f32)
#pragma unroll
        for (int q = 0; q < 16; ++q) {
            int idx  = lane + 32 * q;          // 0..511 of the 32x16 tile
            int row  = idx >> 4;
            int cl   = idx & 15;
            int gcol = nt * 16 + cl;
            float ewv = sh_ew[w][row][cl];
            if (gcol < 256) {                  // w1: out_s[o] += C1*s0*xs[i]*w
                int i = gcol >> 4, o = gcol & 15;
                atomicAdd(&sh_msg[w][row][o], C1c * sh_s0[w][row] * sh_xs[w][row][i] * ewv);
            } else if (gcol < 384) {           // w2: out_v[o][x] += C2*xs[i]*s1[x]*w
                int cb = gcol - 256; int i = cb >> 3, o = cb & 7;
                float cc = C2c * sh_xs[w][row][i] * ewv;
#pragma unroll
                for (int xx = 0; xx < 3; ++xx)
                    atomicAdd(&sh_msg[w][row][16 + o*3 + xx], cc * sh_s1[w][row][xx]);
            } else if (gcol < 448) {           // w3: out_v[o][x] += C3*s0*xv[i][x]*w
                int cb = gcol - 384; int i = cb >> 3, o = cb & 7;
                float cc = C3c * sh_s0[w][row] * ewv;
#pragma unroll
                for (int xx = 0; xx < 3; ++xx)
                    atomicAdd(&sh_msg[w][row][16 + o*3 + xx], cc * sh_xv[w][row][i*3 + xx]);
            } else {                           // w4: out_s[o] += C4*(xv[i].s1)*w
                int cb = gcol - 448; int i = cb >> 4, o = cb & 15;
                atomicAdd(&sh_msg[w][row][o], C4c * sh_vd[w][row][i] * ewv);
            }
        }
        __builtin_amdgcn_wave_barrier();
    }

    // ---- scatter messages to destination nodes (global_atomic_add_f32)
#pragma unroll
    for (int q = 0; q < 40; ++q) {
        int idx = lane + 32 * q;               // 0..1279
        int row = idx / 40;
        int ch  = idx % 40;
        int dn  = sh_dst[w][row];
        atomicAdd(&sums[((size_t)t * N_NODES + dn) * C_OUT + ch], sh_msg[w][row][ch]);
    }
}

// ---------------------------------------------------------------------------
// BN pass 1: mean-divide + spectral add (in place), grouped partial stats.
// Group index g = (c*8 + t) % 40 (faithful to the reference's reshape).
// ---------------------------------------------------------------------------
__global__ void bn_partial(float* __restrict__ y, const float* __restrict__ invc,
                           const float* __restrict__ xsc,
                           float* __restrict__ gsum, float* __restrict__ gsq) {
    __shared__ float ps[40], pq[40];
    int tid = threadIdx.x;
    if (tid < 40) { ps[tid] = 0.f; pq[tid] = 0.f; }
    __syncthreads();
    int idx = blockIdx.x * 256 + tid;          // grid exactly covers T*N*40
    int c = idx % 40;
    int n = (idx / 40) % N_NODES;
    int t = idx / (40 * N_NODES);
    float v = y[idx] * invc[n];
    if (c < 16) v += xsc[((size_t)t * N_NODES + n) * S_OUTc + c];
    y[idx] = v;
    int g = (c * 8 + t) % 40;
    atomicAdd(&ps[g], v);
    atomicAdd(&pq[g], v * v);
    __syncthreads();
    if (tid < 40) { atomicAdd(&gsum[tid], ps[tid]); atomicAdd(&gsq[tid], pq[tid]); }
}

__global__ void bn_finalize(const float* __restrict__ gsum, const float* __restrict__ gsq,
                            const float* __restrict__ gamma, const float* __restrict__ beta,
                            float* __restrict__ scale, float* __restrict__ shift) {
    int g = threadIdx.x;
    if (g < 40) {
        float cnt  = 8.0f * (float)N_NODES;
        float mean = gsum[g] / cnt;
        float var  = gsq[g] / cnt - mean * mean;
        float sc   = gamma[g] * rsqrtf(var + BN_EPS);
        scale[g] = sc;
        shift[g] = beta[g] - mean * sc;
    }
}

__global__ void bn_apply(const float* __restrict__ y, const float* __restrict__ scale,
                         const float* __restrict__ shift, float* __restrict__ out) {
    int idx = blockIdx.x * 256 + threadIdx.x;
    int c = idx % 40;
    int t = idx / (40 * N_NODES);
    int g = (c * 8 + t) % 40;
    out[idx] = y[idx] * scale[g] + shift[g];
}

// ---------------------------------------------------------------------------
extern "C" void kernel_launch(void* const* d_in, const int* in_sizes, int n_in,
                              void* d_out, int out_size, void* d_ws, size_t ws_size,
                              hipStream_t stream) {
    (void)in_sizes; (void)n_in; (void)out_size; (void)ws_size;
    const float* x     = (const float*)d_in[0];
    const int*   ei    = (const int*)  d_in[1];
    const float* ea    = (const float*)d_in[2];
    const float* esh   = (const float*)d_in[3];
    const float* wr    = (const float*)d_in[4];
    const float* wi    = (const float*)d_in[5];
    const float* w1    = (const float*)d_in[6];
    const float* b1    = (const float*)d_in[7];
    const float* w2    = (const float*)d_in[8];
    const float* b2    = (const float*)d_in[9];
    const float* gamma = (const float*)d_in[10];
    const float* beta  = (const float*)d_in[11];

    char* ws = (char*)d_ws;
    size_t off = 0;
    auto walloc = [&](size_t bytes) -> void* {
        void* p = ws + off;
        off = (off + bytes + 255) & ~(size_t)255;
        return p;
    };
    float*    sums  = (float*)   walloc(sizeof(float) * (size_t)T_DIM * N_NODES * C_OUT);
    float*    xsc   = (float*)   walloc(sizeof(float) * (size_t)T_DIM * N_NODES * S_OUTc);
    float*    invc  = (float*)   walloc(sizeof(float) * N_NODES);
    float*    gsum  = (float*)   walloc(sizeof(float) * 40);
    float*    gsq   = (float*)   walloc(sizeof(float) * 40);
    float*    scale = (float*)   walloc(sizeof(float) * 40);
    float*    shift = (float*)   walloc(sizeof(float) * 40);
    _Float16* w1p   = (_Float16*)walloc(sizeof(_Float16) * 4  * 2 * 32 * 16);
    _Float16* w2p   = (_Float16*)walloc(sizeof(_Float16) * 36 * 2 * 32 * 16);

    hipMemsetAsync(sums, 0, sizeof(float) * (size_t)T_DIM * N_NODES * C_OUT, stream);
    hipMemsetAsync(invc, 0, sizeof(float) * N_NODES, stream);
    hipMemsetAsync(gsum, 0, sizeof(float) * 40, stream);
    hipMemsetAsync(gsq,  0, sizeof(float) * 40, stream);

    prep_weights   <<<(36 * 2 * 32 * 16 + 255) / 256, 256, 0, stream>>>(w1, w2, w1p, w2p);
    spectral_kernel<<<N_NODES / 16, 256, 0, stream>>>(x, wr, wi, xsc);
    count_kernel   <<<(E_EDGES + 255) / 256, 256, 0, stream>>>(ei, invc);
    invc_kernel    <<<(N_NODES + 255) / 256, 256, 0, stream>>>(invc);

    // 10000 wave-tiles (T*E/32), 2 waves per 64-thread block
    edge_kernel<<<(T_DIM * E_EDGES / 32) / 2, 64, 0, stream>>>(
        x, ei, ea, esh, b1, b2, w1p, w2p, sums);

    bn_partial <<<(T_DIM * N_NODES * C_OUT) / 256, 256, 0, stream>>>(sums, invc, xsc, gsum, gsq);
    bn_finalize<<<1, 64, 0, stream>>>(gsum, gsq, gamma, beta, scale, shift);
    bn_apply   <<<(T_DIM * N_NODES * C_OUT) / 256, 256, 0, stream>>>(sums, scale, shift, (float*)d_out);
}
